// DDAMEncoder_61916248539183
// MI455X (gfx1250) — matrix operand order repaired and checked by reading, not compiled, not run on hardware
//
#include <hip/hip_runtime.h>
#include <math.h>

typedef _Float16 half_t;
typedef __attribute__((ext_vector_type(16))) _Float16 v16h;
typedef __attribute__((ext_vector_type(8)))  _Float16 v8h;
typedef __attribute__((ext_vector_type(4)))  float    v4f;
typedef __attribute__((ext_vector_type(8)))  float    v8f;

__device__ __forceinline__ v8f v8zero() { v8f z = {0.f,0.f,0.f,0.f,0.f,0.f,0.f,0.f}; return z; }

// ---- fragment loaders --------------------------------------------------------
// A fragment (16x32 f16): lane L: row m=L%16, h=L/16; elems 0..7 -> K=8h+e, 8..15 -> K=16+8h+e
__device__ __forceinline__ v16h frag_a_h(const half_t* A, int lda) {
  int lane = threadIdx.x & 31;
  int m = lane & 15, hh = lane >> 4;
  const half_t* p = A + m * lda + 8 * hh;
  v8h lo = *(const v8h*)(p);
  v8h hi = *(const v8h*)(p + 16);
  return __builtin_shufflevector(lo, hi, 0,1,2,3,4,5,6,7,8,9,10,11,12,13,14,15);
}

__device__ __forceinline__ v16h frag_a_f32(const float* A, int lda) {
  int lane = threadIdx.x & 31;
  int m = lane & 15, hh = lane >> 4;
  const float* p = A + (size_t)m * lda + 8 * hh;
  v4f f0 = *(const v4f*)(p);
  v4f f1 = *(const v4f*)(p + 4);
  v4f f2 = *(const v4f*)(p + 16);
  v4f f3 = *(const v4f*)(p + 20);
  v16h a;
#pragma unroll
  for (int e = 0; e < 4; ++e) {
    a[e]    = (half_t)f0[e];
    a[4+e]  = (half_t)f1[e];
    a[8+e]  = (half_t)f2[e];
    a[12+e] = (half_t)f3[e];
  }
  return a;
}

// B fragment from swizzled storage: block of 32 lanes x 16 halfs, lane-contiguous
__device__ __forceinline__ v16h frag_b_sw(const half_t* Bsw) {
  int lane = threadIdx.x & 31;
  return *(const v16h*)(Bsw + lane * 16);
}

__device__ __forceinline__ v8f wmma16(v16h a, v16h b, v8f c) {
  return __builtin_amdgcn_wmma_f32_16x16x32_f16(false, a, false, b, (short)0, c, false, false);
}

// swizzled index for B weight element (k,n), NT = N/16 tiles
__device__ __forceinline__ int swidx(int k, int n, int NT) {
  int lane = (((k >> 3) & 1) << 4) | (n & 15);
  return (((k >> 5) * NT + (n >> 4)) << 9) + (lane << 4) + (((k >> 4) & 1) << 3) + (k & 7);
}

__constant__ int c_IDX[16]  = {0, 7, 10, 13, 1, 4, 11, 14, 2, 5, 8, 15, 3, 6, 9, 12};
__constant__ int c_BACK[16] = {0, 4, 8, 12, 5, 9, 13, 1, 10, 14, 2, 6, 15, 3, 7, 11};

// ---------------------------------------------------------------- weight prep
__global__ void k_prep(const float* __restrict__ wqkv, const float* __restrict__ wout,
                       const float* __restrict__ w1,  const float* __restrict__ w2,
                       const float* __restrict__ c3,  const float* __restrict__ c4,
                       half_t* WQ, half_t* WO, half_t* W1, half_t* W2,
                       half_t* B3, half_t* B9, half_t* Cm, half_t* Sm)
{
  int i = blockIdx.x * 256 + threadIdx.x;
  const int nWQ = 2*256*128, nWO = 2*32*256, nW1 = 2*256*512, nW2 = 2*512*256;
  const int nB3 = 512*256, nB9 = 9*256*256, nT = 4096;
  if (i < nWQ) { int dep = i >> 15, r = i & 32767, k = r >> 7, n = r & 127;
                 float v = (n < 96) ? wqkv[dep*24576 + k*96 + n] : 0.f;
                 WQ[dep*32768 + swidx(k, n, 8)] = (half_t)v; return; } i -= nWQ;
  if (i < nWO) { int dep = i / 8192, r = i % 8192, k = r >> 8, n = r & 255;
                 WO[dep*8192 + swidx(k, n, 16)] = (half_t)wout[i]; return; } i -= nWO;
  if (i < nW1) { int dep = i / 131072, r = i % 131072, k = r >> 9, n = r & 511;
                 W1[dep*131072 + swidx(k, n, 32)] = (half_t)w1[i]; return; } i -= nW1;
  if (i < nW2) { int dep = i / 131072, r = i % 131072, k = r >> 8, n = r & 255;
                 W2[dep*131072 + swidx(k, n, 16)] = (half_t)w2[i]; return; } i -= nW2;
  if (i < nB3) { int ci = i >> 8, co = i & 255;
                 B3[swidx(ci, co, 16)] = (half_t)c3[(size_t)co*512 + ci]; return; } i -= nB3;
  if (i < nB9) { int t = i >> 16, r = i & 65535, ci = r >> 8, co = r & 255;
                 B9[t*65536 + swidx(ci, co, 16)] = (half_t)c4[((size_t)co*256 + ci)*9 + t];
                 return; } i -= nB9;
  if (i < nT)  { int k = i >> 6, n = i & 63;
                 float ang = 6.283185307179586f * (float)((k*n) & 63) / 64.f;
                 Cm[i] = (half_t)(cosf(ang) * 0.125f);
                 Sm[i] = (half_t)(sinf(ang) * 0.125f); }
}

// ---------------------------------------------------------------- unfold/fold
__global__ void k_unfold(const float* __restrict__ x, float* __restrict__ X4)
{
  int i = blockIdx.x * 256 + threadIdx.x;       // g(2) b(4) p(2) l(8) d(8)
  int d = i & 255, l = (i >> 8) & 255, p = (i >> 16) & 3, b = (i >> 18) & 15, g = i >> 22;
  int kpos = c_IDX[g*4 + p];
  int ky = kpos >> 2, kx = kpos & 3;
  int ph = l >> 4, pw = l & 15;
  int h = ph*4 + ky, w = pw*4 + kx;
  X4[i] = x[(((size_t)b*256 + d)*64 + h)*64 + w];
}

__global__ void k_fold(const float* __restrict__ X4, float* __restrict__ zf)
{
  int i = blockIdx.x * 256 + threadIdx.x;       // b(4) h(6) w(6) d(8)
  int d = i & 255, w = (i >> 8) & 63, h = (i >> 14) & 63, b = i >> 20;
  int kpos = (h & 3)*4 + (w & 3);
  int s = c_BACK[kpos];
  int g = s >> 2, p = s & 3;
  int l = (h >> 2)*16 + (w >> 2);
  zf[i] = X4[((((size_t)g*16 + b)*4 + p)*256 + l)*256 + d];
}

// ---------------------------------------------------------------- elementwise
__global__ void k_add(float* __restrict__ d, const float* __restrict__ s)
{ int i = blockIdx.x*256 + threadIdx.x; d[i] += s[i]; }
__global__ void k_copy(float* __restrict__ d, const float* __restrict__ s)
{ int i = blockIdx.x*256 + threadIdx.x; d[i] = s[i]; }

// ---------------------------------------------------------------- layernorm (dim 256)
__global__ __launch_bounds__(256) void k_ln(const float* __restrict__ X,
    const float* __restrict__ g, const float* __restrict__ b, float* __restrict__ O)
{
  __shared__ float s1[256], s2[256];
  int row = blockIdx.x, t = threadIdx.x;
  float v = X[(size_t)row*256 + t];
  s1[t] = v; s2[t] = v*v;
  __syncthreads();
  for (int s = 128; s > 0; s >>= 1) {
    if (t < s) { s1[t] += s1[t+s]; s2[t] += s2[t+s]; }
    __syncthreads();
  }
  float mean = s1[0] * (1.f/256.f);
  float var  = s2[0] * (1.f/256.f) - mean*mean;
  O[(size_t)row*256 + t] = (v - mean) * rsqrtf(var + 1e-5f) * g[t] + b[t];
}

// ---------------------------------------------------------------- WMMA GEMM (LDS-free)
// C[M,N] = act(A[M,K]f32 @ Bsw[K,N]f16 + bias) + resid ; N multiple of 64, K multiple of 32
__global__ __launch_bounds__(128) void k_gemm(
    const float* __restrict__ A, const half_t* __restrict__ Bsw,
    const float* __restrict__ bias, const float* __restrict__ resid,
    float* __restrict__ C, int M, int N, int K, int act)
{
  const int m0 = blockIdx.x * 64, n0t = blockIdx.y * 4;
  const int wave = threadIdx.x >> 5, lane = threadIdx.x & 31;
  const int NT = N >> 4;
  v8f acc0 = v8zero(), acc1 = v8zero(), acc2 = v8zero(), acc3 = v8zero();
  const float* Arow = A + (size_t)(m0 + wave*16) * K;
  const half_t* Bbase = Bsw + ((size_t)n0t << 9);
  const int kcN = K >> 5;
#pragma unroll 2
  for (int kc = 0; kc < kcN; ++kc) {
    v16h a = frag_a_f32(Arow + kc*32, K);
    const half_t* Bp = Bbase + (((size_t)kc * NT) << 9);
    v16h b0 = frag_b_sw(Bp);
    v16h b1 = frag_b_sw(Bp + 512);
    v16h b2 = frag_b_sw(Bp + 1024);
    v16h b3 = frag_b_sw(Bp + 1536);
    acc0 = wmma16(a, b0, acc0);
    acc1 = wmma16(a, b1, acc1);
    acc2 = wmma16(a, b2, acc2);
    acc3 = wmma16(a, b3, acc3);
  }
  const int lm = 8*(lane >> 4), ln = lane & 15;
  v8f accs[4] = {acc0, acc1, acc2, acc3};
  for (int nt = 0; nt < 4; ++nt) {
    int gn = (n0t + nt)*16 + ln;
    float bv = bias ? bias[gn] : 0.f;
    for (int r = 0; r < 8; ++r) {
      int gm = m0 + wave*16 + lm + r;
      float v = accs[nt][r] + bv;
      if (act == 1) v = v / (1.f + __expf(-v));
      if (resid) v += resid[(size_t)gm * N + gn];
      C[(size_t)gm * N + gn] = v;
    }
  }
}

// ---------------------------------------------------------------- attention
// qkv rows have stride 128 (cols 0..31 q, 32..63 k, 64..95 v, 96..127 pad)
__global__ __launch_bounds__(64) void k_attn(const float* __restrict__ qkv,
                                             float* __restrict__ ao)
{
  int bid = blockIdx.x;                 // 64 * 4 * 8
  int chunk = bid & 7, head = (bid >> 3) & 3, bp = bid >> 5;
  __shared__ alignas(32) half_t Qs[32*32];      // row-major (frag_a)
  __shared__ alignas(32) half_t Ksw[16*32*16];  // swizzled: 16 n-tiles x 1 k-chunk
  __shared__ alignas(32) half_t Vsw[8*32*16];   // swizzled: 8 k-chunks x 1 n-tile
  __shared__ alignas(32) half_t Ss[32*256];     // row-major scores / exp(P)
  __shared__ float rs[32];
  int tid = threadIdx.x, wave = tid >> 5, lane = tid & 31;
  size_t base = (size_t)bp * 256 * 128;
  int q0 = chunk * 32;
  for (int e = tid; e < 32*32; e += 64) {
    int r = e >> 5, j = e & 31;
    Qs[e] = (j < 8) ? (half_t)qkv[base + (size_t)(q0 + r)*128 + head*8 + j] : (half_t)0.f;
  }
  for (int e = tid; e < 16*32*16; e += 64) {    // B = K^T [32 j][256 tok]
    int nt = e >> 9, ln2 = (e >> 4) & 31, el = e & 15;
    int j = ((el >> 3) << 4) + ((ln2 >> 4) << 3) + (el & 7);
    int tok = nt*16 + (ln2 & 15);
    Ksw[e] = (j < 8) ? (half_t)qkv[base + (size_t)tok*128 + 32 + head*8 + j] : (half_t)0.f;
  }
  for (int e = tid; e < 8*32*16; e += 64) {     // B = V [256 tok][16 j]
    int kc = e >> 9, ln2 = (e >> 4) & 31, el = e & 15;
    int tok = kc*32 + ((el >> 3) << 4) + ((ln2 >> 4) << 3) + (el & 7);
    int j = ln2 & 15;
    Vsw[e] = (j < 8) ? (half_t)qkv[base + (size_t)tok*128 + 64 + head*8 + j] : (half_t)0.f;
  }
  __syncthreads();
  {
    v16h a = frag_a_h(Qs + wave*16*32, 32);
    int lm = 8*(lane >> 4), ln = lane & 15;
#pragma unroll
    for (int nt = 0; nt < 16; ++nt) {
      v8f acc = v8zero();
      acc = wmma16(a, frag_b_sw(Ksw + nt*512), acc);
#pragma unroll
      for (int r = 0; r < 8; ++r) {
        int m = wave*16 + lm + r;
        Ss[m*256 + nt*16 + ln] = (half_t)(acc[r] * 0.3535533906f);
      }
    }
  }
  __syncthreads();
  if (tid < 32) {                        // unnormalized softmax; keep 1/sum
    float mx = -1e30f;
    for (int c = 0; c < 256; ++c) mx = fmaxf(mx, (float)Ss[tid*256 + c]);
    float sum = 0.f;
    for (int c = 0; c < 256; ++c) {
      float v = __expf((float)Ss[tid*256 + c] - mx);
      sum += v; Ss[tid*256 + c] = (half_t)v;
    }
    rs[tid] = 1.f / sum;
  }
  __syncthreads();
  {
    v8f acc = v8zero();
#pragma unroll
    for (int kc = 0; kc < 8; ++kc) {
      v16h a = frag_a_h(Ss + wave*16*256 + kc*32, 256);
      acc = wmma16(a, frag_b_sw(Vsw + kc*512), acc);
    }
    int lm = 8*(lane >> 4), ln = lane & 15;
    if (ln < 8) {
#pragma unroll
      for (int r = 0; r < 8; ++r) {
        int m = wave*16 + lm + r;
        ao[((size_t)bp*256 + q0 + m)*32 + head*8 + ln] = acc[r] * rs[m];
      }
    }
  }
}

// ---------------------------------------------------------------- DFT stage (WMMA)
// mode 0: Re=CR, Im=-SR (fwd, real in); 1: Re=CR+SI, Im=CI-SR (fwd); 2: Re=CR-SI, Im=CI+SR (inv)
__global__ __launch_bounds__(128) void k_dft(
    const void* inRe_, const void* inIm_, void* outRe_, void* outIm_,
    const half_t* __restrict__ Cg, const half_t* __restrict__ Sg,
    long strideB, long strideS, long strideI, int mode, int inF32, int outF32)
{
  __shared__ alignas(32) half_t Xr[8*32*16];   // swizzled: (kc,nt) blocks
  __shared__ alignas(32) half_t Xi[8*32*16];
  int sl = blockIdx.x;
  int n0 = blockIdx.y * 64;
  long base = (long)(sl >> 6) * strideB + (long)(sl & 63) * strideS;
  int tid = threadIdx.x, wave = tid >> 5, lane = tid & 31;
  bool hasIm = (inIm_ != nullptr);
  for (int e = tid; e < 4096; e += 128) {
    int kc = (e >> 11) & 1, nt = (e >> 9) & 3, ln2 = (e >> 4) & 31, el = e & 15;
    int i = kc*32 + ((el >> 3) << 4) + ((ln2 >> 4) << 3) + (el & 7);     // spatial index (K dim)
    int dd = nt*16 + (ln2 & 15);                // channel within 64 (N dim)
    long idx = base + (long)i * strideI + n0 + dd;
    Xr[e] = inF32 ? (half_t)((const float*)inRe_)[idx] : ((const half_t*)inRe_)[idx];
    Xi[e] = hasIm ? ((const half_t*)inIm_)[idx] : (half_t)0.f;
  }
  v16h aC0 = frag_a_h(Cg + wave*16*64,      64);
  v16h aC1 = frag_a_h(Cg + wave*16*64 + 32, 64);
  v16h aS0 = frag_a_h(Sg + wave*16*64,      64);
  v16h aS1 = frag_a_h(Sg + wave*16*64 + 32, 64);
  __syncthreads();
  int lm = 8*(lane >> 4), ln = lane & 15;
#pragma unroll
  for (int ns = 0; ns < 4; ++ns) {
    v8f cr = v8zero(), sr = v8zero(), ci = v8zero(), si = v8zero();
    v16h br0 = frag_b_sw(Xr + ns*512);
    v16h br1 = frag_b_sw(Xr + (4 + ns)*512);
    v16h bi0 = frag_b_sw(Xi + ns*512);
    v16h bi1 = frag_b_sw(Xi + (4 + ns)*512);
    cr = wmma16(aC0, br0, cr); cr = wmma16(aC1, br1, cr);
    sr = wmma16(aS0, br0, sr); sr = wmma16(aS1, br1, sr);
    ci = wmma16(aC0, bi0, ci); ci = wmma16(aC1, bi1, ci);
    si = wmma16(aS0, bi0, si); si = wmma16(aS1, bi1, si);
    v8f oRe, oIm;
    if (mode == 0)      { oRe = cr;      oIm = v8zero() - sr; }
    else if (mode == 1) { oRe = cr + si; oIm = ci - sr; }
    else                { oRe = cr - si; oIm = ci + sr; }
#pragma unroll
    for (int r = 0; r < 8; ++r) {
      int m = wave*16 + lm + r;
      long idx = base + (long)m * strideI + n0 + ns*16 + ln;
      if (outF32) ((float*)outRe_)[idx] = oRe[r];
      else        ((half_t*)outRe_)[idx] = (half_t)oRe[r];
      if (outIm_) ((half_t*)outIm_)[idx] = (half_t)oIm[r];
    }
  }
}

// spectral weight multiply, Hermitian extension for kw>32
__global__ void k_fmul(half_t* __restrict__ Re, half_t* __restrict__ Im,
                       const float* __restrict__ W)
{
  int i = blockIdx.x * 256 + threadIdx.x;    // b(4) kh(6) kw(6) d(8)
  int d = i & 255, kw = (i >> 8) & 63, kh = (i >> 14) & 63;
  float wr, wi;
  if (kw <= 32) {
    const float* p = W + (((size_t)kh*33 + kw)*256 + d)*2;
    wr = p[0]; wi = p[1];
  } else {
    int kh2 = (64 - kh) & 63, kw2 = 64 - kw;
    const float* p = W + (((size_t)kh2*33 + kw2)*256 + d)*2;
    wr = p[0]; wi = -p[1];
  }
  float re = (float)Re[i], im = (float)Im[i];
  Re[i] = (half_t)(re*wr - im*wi);
  Im[i] = (half_t)(re*wi + im*wr);
}

// ---------------------------------------------------------------- conv3 (1x1, 512->256) + BN + SiLU
__global__ __launch_bounds__(128) void k_conv3(
    const float* __restrict__ zf, const float* __restrict__ x,
    const half_t* __restrict__ B3,
    const float* __restrict__ g, const float* __restrict__ bb,
    const float* __restrict__ mn, const float* __restrict__ vr,
    float* __restrict__ out)
{
  int bi = blockIdx.x >> 6, mt = blockIdx.x & 63;
  int n0t = blockIdx.y * 4, m0 = mt * 64;
  int wave = threadIdx.x >> 5, lane = threadIdx.x & 31;
  v8f acc0 = v8zero(), acc1 = v8zero(), acc2 = v8zero(), acc3 = v8zero();
  const int gmb = m0 + wave*16;
  const float* zrow = zf + ((size_t)bi*4096 + gmb)*256;
  for (int kc = 0; kc < 16; ++kc) {
    int k0 = kc*32;
    v16h a;
    if (k0 < 256) {
      a = frag_a_f32(zrow + k0, 256);
    } else {
      int m = lane & 15, hh = lane >> 4;
      const float* xp = x + (size_t)bi*256*4096 + (gmb + m);
      int kb = k0 - 256 + 8*hh;
#pragma unroll
      for (int e = 0; e < 8; ++e) {
        a[e]   = (half_t)xp[(size_t)(kb + e)*4096];
        a[8+e] = (half_t)xp[(size_t)(kb + 16 + e)*4096];
      }
    }
    const half_t* Bp = B3 + (((size_t)(kc*16 + n0t)) << 9);
    v16h b0 = frag_b_sw(Bp);
    v16h b1 = frag_b_sw(Bp + 512);
    v16h b2 = frag_b_sw(Bp + 1024);
    v16h b3 = frag_b_sw(Bp + 1536);
    acc0 = wmma16(a, b0, acc0);
    acc1 = wmma16(a, b1, acc1);
    acc2 = wmma16(a, b2, acc2);
    acc3 = wmma16(a, b3, acc3);
  }
  int lm = 8*(lane >> 4), ln = lane & 15;
  v8f accs[4] = {acc0, acc1, acc2, acc3};
  for (int nt = 0; nt < 4; ++nt) {
    int c = (n0t + nt)*16 + ln;
    float s  = g[c] * rsqrtf(vr[c] + 1e-5f);
    float sh = bb[c] - mn[c]*s;
    for (int r = 0; r < 8; ++r) {
      int hw = m0 + wave*16 + lm + r;
      float v = accs[nt][r]*s + sh;
      v = v / (1.f + __expf(-v));
      out[((size_t)bi*4096 + hw)*256 + c] = v;
    }
  }
}

// ---------------------------------------------------------------- conv4 (3x3, 256->256) + BN + SiLU
__global__ __launch_bounds__(128) void k_conv4(
    const float* __restrict__ in, const half_t* __restrict__ B9,
    const float* __restrict__ g, const float* __restrict__ bb,
    const float* __restrict__ mn, const float* __restrict__ vr,
    float* __restrict__ out)
{
  int bi = blockIdx.x >> 6, h = blockIdx.x & 63;
  int n0t = blockIdx.y * 4;
  int wave = threadIdx.x >> 5, lane = threadIdx.x & 31;
  int m = lane & 15, hh = lane >> 4;
  v8f acc0 = v8zero(), acc1 = v8zero(), acc2 = v8zero(), acc3 = v8zero();
  for (int t = 0; t < 9; ++t) {
    int dy = t/3 - 1, dx = t%3 - 1;
    int hy = h + dy;
    if (hy < 0 || hy >= 64) continue;
    int wx = wave*16 + m + dx;
    bool ok = (wx >= 0 && wx < 64);
    const float* p = in + (((size_t)bi*64 + hy)*64 + wx)*256;
    for (int kc = 0; kc < 8; ++kc) {
      v16h a;
      if (ok) {
        const float* q = p + kc*32 + 8*hh;
        v4f f0 = *(const v4f*)(q);
        v4f f1 = *(const v4f*)(q + 4);
        v4f f2 = *(const v4f*)(q + 16);
        v4f f3 = *(const v4f*)(q + 20);
#pragma unroll
        for (int e = 0; e < 4; ++e) {
          a[e] = (half_t)f0[e]; a[4+e] = (half_t)f1[e];
          a[8+e] = (half_t)f2[e]; a[12+e] = (half_t)f3[e];
        }
      } else {
#pragma unroll
        for (int e = 0; e < 16; ++e) a[e] = (half_t)0.f;
      }
      const half_t* Bp = B9 + (size_t)t*65536 + (((size_t)(kc*16 + n0t)) << 9);
      v16h b0 = frag_b_sw(Bp);
      v16h b1 = frag_b_sw(Bp + 512);
      v16h b2 = frag_b_sw(Bp + 1024);
      v16h b3 = frag_b_sw(Bp + 1536);
      acc0 = wmma16(a, b0, acc0);
      acc1 = wmma16(a, b1, acc1);
      acc2 = wmma16(a, b2, acc2);
      acc3 = wmma16(a, b3, acc3);
    }
  }
  int lm = 8*(lane >> 4), ln = lane & 15;
  v8f accs[4] = {acc0, acc1, acc2, acc3};
  for (int nt = 0; nt < 4; ++nt) {
    int c = (n0t + nt)*16 + ln;
    float s  = g[c] * rsqrtf(vr[c] + 1e-5f);
    float sh = bb[c] - mn[c]*s;
    for (int r = 0; r < 8; ++r) {
      int w = wave*16 + lm + r;
      float v = accs[nt][r]*s + sh;
      v = v / (1.f + __expf(-v));
      out[(((size_t)bi*256 + c)*64 + h)*64 + w] = v;
    }
  }
}

// ================================================================= launcher
extern "C" void kernel_launch(void* const* d_in, const int* in_sizes, int n_in,
                              void* d_out, int out_size, void* d_ws, size_t ws_size,
                              hipStream_t stream) {
  (void)in_sizes; (void)n_in; (void)out_size; (void)ws_size;
  const float* x    = (const float*)d_in[0];
  const float* ln1g = (const float*)d_in[1];
  const float* ln1b = (const float*)d_in[2];
  const float* wqkv = (const float*)d_in[3];
  const float* wout = (const float*)d_in[4];
  const float* bout = (const float*)d_in[5];
  const float* ln2g = (const float*)d_in[6];
  const float* ln2b = (const float*)d_in[7];
  const float* w1   = (const float*)d_in[8];
  const float* b1   = (const float*)d_in[9];
  const float* w2   = (const float*)d_in[10];
  const float* b2   = (const float*)d_in[11];
  const float* fftw = (const float*)d_in[12];
  const float* c3w  = (const float*)d_in[13];
  const float* bn3g = (const float*)d_in[14];
  const float* bn3b = (const float*)d_in[15];
  const float* bn3m = (const float*)d_in[16];
  const float* bn3v = (const float*)d_in[17];
  const float* c4w  = (const float*)d_in[18];
  const float* bn4g = (const float*)d_in[19];
  const float* bn4b = (const float*)d_in[20];
  const float* bn4m = (const float*)d_in[21];
  const float* bn4v = (const float*)d_in[22];

  const size_t MB = 1024ull*1024ull;
  char* ws = (char*)d_ws;
  float*  X4   = (float*)(ws + 0);              // 4 x 16MB activation groups
  float*  Aacc = (float*)(ws + 64*MB);          // 16MB
  float*  hbuf = (float*)(ws + 80*MB);          // 16MB
  float*  qkvb = (float*)(ws + 96*MB);          // 8MB (16384 x 128 f32)
  float*  aob  = (float*)(ws + 104*MB);         // 4MB
  float*  mhb  = (float*)(ws + 108*MB);         // 32MB  (ends 140MB)
  half_t* Re1  = (half_t*)(ws + 64*MB);         // 32MB f16 (overlays dead tf temps)
  half_t* Im1  = (half_t*)(ws + 96*MB);         // 32MB f16 (overlays dead tf temps)
  float*  zf   = (float*)(ws + 140*MB);         // 64MB (ends 204MB)
  half_t* Re2  = (half_t*)(ws + 204*MB);        // 32MB
  half_t* Im2  = (half_t*)(ws + 236*MB);        // 32MB (ends 268MB)
  half_t* WQ   = (half_t*)(ws + 268*MB);
  half_t* WO   = WQ  + 2*256*128;
  half_t* W1h  = WO  + 2*32*256;
  half_t* W2h  = W1h + 2*256*512;
  half_t* B3   = W2h + 2*512*256;
  half_t* B9   = B3  + 512*256;
  half_t* Cm   = B9  + 9*256*256;
  half_t* Sm   = Cm  + 4096;
  float*  out1 = (float*)(ws + 0);              // conv3 result reuses X4 region

  k_prep<<<5200, 256, 0, stream>>>(wqkv, wout, w1, w2, c3w, c4w,
                                   WQ, WO, W1h, W2h, B3, B9, Cm, Sm);
  k_unfold<<<65536, 256, 0, stream>>>(x, X4);

  const int NG = 4194304;
  const int NB = NG / 256;
  float* Xg[4] = {X4, X4 + NG, X4 + 2*NG, X4 + 3*NG};

  auto tf = [&](float* X) {
    for (int dep = 0; dep < 2; ++dep) {
      k_ln<<<16384, 256, 0, stream>>>(X, ln1g + dep*256, ln1b + dep*256, hbuf);
      k_gemm<<<dim3(256, 2), 128, 0, stream>>>(hbuf, WQ + dep*32768, nullptr, nullptr,
                                               qkvb, 16384, 128, 256, 0);
      k_attn<<<2048, 64, 0, stream>>>(qkvb, aob);
      k_gemm<<<dim3(256, 4), 128, 0, stream>>>(aob, WO + dep*8192, bout + dep*256, X,
                                               X, 16384, 256, 32, 0);
      k_ln<<<16384, 256, 0, stream>>>(X, ln2g + dep*256, ln2b + dep*256, hbuf);
      k_gemm<<<dim3(256, 8), 128, 0, stream>>>(hbuf, W1h + dep*131072, b1 + dep*512, nullptr,
                                               mhb, 16384, 512, 256, 1);
      k_gemm<<<dim3(256, 4), 128, 0, stream>>>(mhb, W2h + dep*131072, b2 + dep*256, X,
                                               X, 16384, 256, 512, 0);
    }
  };

  tf(Xg[0]);                                        // Xg0 = x1
  k_copy<<<NB, 256, 0, stream>>>(Aacc, Xg[0]);      // A = x1
  k_add <<<NB, 256, 0, stream>>>(Xg[1], Aacc); tf(Xg[1]);   // Xg1 = x2
  k_add <<<NB, 256, 0, stream>>>(Aacc, Xg[1]);      // A = x1+x2
  k_add <<<NB, 256, 0, stream>>>(Xg[2], Aacc); tf(Xg[2]);   // Xg2 = x3
  k_add <<<NB, 256, 0, stream>>>(Aacc, Xg[2]);      // A = x1+x2+x3
  k_add <<<NB, 256, 0, stream>>>(Xg[3], Aacc); tf(Xg[3]);   // Xg3 = x4

  k_fold<<<65536, 256, 0, stream>>>(X4, zf);

  const long SB = 1048576;  // per-batch stride (64*64*256)
  k_dft<<<dim3(1024, 4), 128, 0, stream>>>(zf, nullptr, Re1, Im1, Cm, Sm,
                                           SB, 16384, 256, 0, 1, 0);
  k_dft<<<dim3(1024, 4), 128, 0, stream>>>(Re1, Im1, Re2, Im2, Cm, Sm,
                                           SB, 256, 16384, 1, 0, 0);
  k_fmul<<<65536, 256, 0, stream>>>(Re2, Im2, fftw);
  k_dft<<<dim3(1024, 4), 128, 0, stream>>>(Re2, Im2, Re1, Im1, Cm, Sm,
                                           SB, 256, 16384, 2, 0, 0);
  k_dft<<<dim3(1024, 4), 128, 0, stream>>>(Re1, Im1, zf, nullptr, Cm, Sm,
                                           SB, 16384, 256, 2, 0, 1);

  k_conv3<<<dim3(1024, 4), 128, 0, stream>>>(zf, x, B3, bn3g, bn3b, bn3m, bn3v, out1);
  k_conv4<<<dim3(1024, 4), 128, 0, stream>>>(out1, B9, bn4g, bn4b, bn4m, bn4v,
                                             (float*)d_out);
}
